// noiseRNN_37117107372281
// MI455X (gfx1250) — matrix operand order repaired and checked by reading, not compiled, not run on hardware
//
#include <hip/hip_runtime.h>
#include <hip/hip_bf16.h>
#include <type_traits>

// noiseRNN scan for MI455X (gfx1250), wave32 + v_wmma_f32_16x16x32_bf16.
//
//   T=2048 sequential steps; per step, per workgroup (16 batch rows):
//     h_new[16,128] = tanh( x_t[16,64]@W_ih^T + h[16,128]@W_hh^T + b )
//     out_t = h_new ; h = h_new + 0.1*noise_t
//
//   - recurrence independent per batch row -> 16 WGs, no cross-WG sync
//   - weights pinned in VGPRs (bf16 B-layout) for the whole loop
//   - hidden state ping-pongs between two LDS buffers -> 1 barrier/step
//   - CDNA5 split barrier (s_barrier_signal -1 / s_barrier_wait -1):
//     next step's x/noise loads + prefetches issue BETWEEN signal and wait
//   - x/noise software-pipelined one step ahead (they don't depend on h)
//   - last iteration peeled -> branch-free hot loop

typedef __attribute__((ext_vector_type(16))) __bf16 v16bf;
typedef __attribute__((ext_vector_type(8)))  __bf16 v8bf;
typedef __attribute__((ext_vector_type(8)))  float  v8f;

#define T_STEPS 2048
#define BATCH   256
#define IN      64
#define HID     128
#define NSTD    0.1f
#define HPAD    136   // LDS row stride (elements); 272B -> 4-bank rotation per row

__device__ __forceinline__ float fast_tanh(float x) {
#if __has_builtin(__builtin_amdgcn_tanhf)
    return __builtin_amdgcn_tanhf(x);            // v_tanh_f32 (CDNA5 trans op)
#else
    float e = __expf(2.0f * x);
    return 1.0f - 2.0f / (e + 1.0f);
#endif
}

// Split workgroup barrier: publish LDS stores, signal, (caller issues
// independent work), then wait. "memory" clobbers keep LDS ops on the
// correct side of each half.
__device__ __forceinline__ void wg_barrier_signal() {
    asm volatile("s_wait_dscnt 0x0\n\ts_barrier_signal -1" ::: "memory");
}
__device__ __forceinline__ void wg_barrier_wait() {
    asm volatile("s_barrier_wait -1" ::: "memory");
}

__device__ __forceinline__ void cvt8(const float4 a, const float4 b, v16bf& dst, int off) {
    dst[off + 0] = (__bf16)a.x; dst[off + 1] = (__bf16)a.y;
    dst[off + 2] = (__bf16)a.z; dst[off + 3] = (__bf16)a.w;
    dst[off + 4] = (__bf16)b.x; dst[off + 5] = (__bf16)b.y;
    dst[off + 6] = (__bf16)b.z; dst[off + 7] = (__bf16)b.w;
}

// A layout (16-bit, 16x32): lane<16 holds K=32c+{0..7,16..23},
// lane>=16 holds K=32c+{8..15,24..31}, row = lane&15.
__device__ __forceinline__ void load_x_tile(const float* __restrict__ xr, int half,
                                            v16bf Ax[2]) {
    #pragma unroll
    for (int c = 0; c < 2; ++c) {
        const int lo = 32 * c + 8 * half;
        const float4 f0 = *(const float4*)(xr + lo);
        const float4 f1 = *(const float4*)(xr + lo + 4);
        const float4 f2 = *(const float4*)(xr + lo + 16);
        const float4 f3 = *(const float4*)(xr + lo + 20);
        cvt8(f0, f1, Ax[c], 0);
        cvt8(f2, f3, Ax[c], 8);
    }
}

__device__ __forceinline__ void load_noise(const float* __restrict__ np, float nz[8]) {
    #pragma unroll
    for (int j = 0; j < 8; ++j) nz[j] = np[(size_t)j * HID];
}

__global__ __launch_bounds__(256, 1) void rnn_scan_kernel(
    const float* __restrict__ x,        // [T,B,I]
    const float* __restrict__ w_ih,     // [H,I]
    const float* __restrict__ w_hh,     // [H,H]
    const float* __restrict__ b_ih,     // [H]
    const float* __restrict__ b_hh,     // [H]
    const float* __restrict__ noise,    // [T,B,H]
    const float* __restrict__ hidden_in,// [1,B,H]
    float* __restrict__ out)            // [T,B,H] ++ [1,B,H]
{
    __shared__ __bf16 Hs[2][16 * HPAD]; // ping-pong hidden state, bf16

    const int tid   = threadIdx.x;
    const int wave  = tid >> 5;         // 0..7 : owns output columns [16w,16w+16)
    const int lane  = tid & 31;
    const int half  = lane >> 4;        // lane-half selects K sub-range (ISA layout)
    const int ln    = lane & 15;
    const int bbase = blockIdx.x * 16;  // batch rows [bbase, bbase+16)
    const int col   = wave * 16 + ln;   // C/D column this lane owns

    // ---- preload B matrices (weights) into VGPRs, f32 -> bf16 --------------
    // B layout (16-bit, 32x16): lane = column, K order matches A,
    // lanes 0-15 hold K=32c+{0..7,16..23}, lanes 16-31 hold K=32c+{8..15,24..31}.
    v16bf Bh[4], Bx[2];
    {
        const float* ph = w_hh + col * HID;       // weight row n == output col
        #pragma unroll
        for (int c = 0; c < 4; ++c) {
            const float* p = ph + 32 * c + 8 * half;
            #pragma unroll
            for (int j = 0; j < 16; ++j) Bh[c][j] = (__bf16)p[(j & 7) + 16 * (j >> 3)];
        }
        const float* pi = w_ih + col * IN;
        #pragma unroll
        for (int c = 0; c < 2; ++c) {
            const float* p = pi + 32 * c + 8 * half;
            #pragma unroll
            for (int j = 0; j < 16; ++j) Bx[c][j] = (__bf16)p[(j & 7) + 16 * (j >> 3)];
        }
    }
    const float bias = b_ih[col] + b_hh[col];

    const int    arow = ln;                        // A row for both lane halves
    const size_t BH   = (size_t)BATCH * HID;

    // ---- pipeline prologue: x/noise for t=0 -------------------------------
    v16bf Ax[2];
    float nz[8];
    load_x_tile(x + (size_t)(bbase + arow) * IN, half, Ax);
    load_noise(noise + (size_t)(bbase + 8 * half) * HID + col, nz);

    // ---- init LDS hidden state (buffer 0) from hidden_in -------------------
    for (int i = tid; i < 16 * HID; i += 256) {
        const int r = i >> 7, cc = i & (HID - 1);
        Hs[0][r * HPAD + cc] = (__bf16)hidden_in[(bbase + r) * HID + cc];
    }
    wg_barrier_signal();

    int p = 0;                                     // current read buffer

    auto step = [&](int t, auto lastc) {
        constexpr bool LAST = decltype(lastc)::value;
        const size_t obase = (size_t)t * BH;

        wg_barrier_wait();                         // h from previous step ready

        // ---- A (h) : 4 K-chunks straight from LDS read buffer --------------
        const __bf16* Hr = Hs[p];
        __bf16*       Hw = Hs[p ^ 1];
        v16bf Ah[4];
        #pragma unroll
        for (int c = 0; c < 4; ++c) {
            const int lo = 32 * c + 8 * half;
            const v8bf h0 = *(const v8bf*)(&Hr[arow * HPAD + lo]);
            const v8bf h1 = *(const v8bf*)(&Hr[arow * HPAD + lo + 16]);
            #pragma unroll
            for (int j = 0; j < 8; ++j) { Ah[c][j] = h0[j]; Ah[c][8 + j] = h1[j]; }
        }

        // ---- 6 chained WMMAs: D = Ax*Bx + Ah*Bh + bias ---------------------
        v8f acc;
        #pragma unroll
        for (int j = 0; j < 8; ++j) acc[j] = bias;
        acc = __builtin_amdgcn_wmma_f32_16x16x32_bf16(false, Ax[0], false, Bx[0],
                                                      (short)0, acc, false, false);
        acc = __builtin_amdgcn_wmma_f32_16x16x32_bf16(false, Ax[1], false, Bx[1],
                                                      (short)0, acc, false, false);
        #pragma unroll
        for (int c = 0; c < 4; ++c)
            acc = __builtin_amdgcn_wmma_f32_16x16x32_bf16(false, Ah[c], false, Bh[c],
                                                          (short)0, acc, false, false);

        // ---- epilogue: tanh, emit out_t, inject noise into carried state ---
        // D layout: lane<16 -> rows 0..7 in VGPR 0..7 ; lane>=16 -> rows 8..15.
        #pragma unroll
        for (int j = 0; j < 8; ++j) {
            const int    m  = j + 8 * half;                 // batch row in tile
            const size_t gi = obase + (size_t)(bbase + m) * HID + col;
            const float  v  = fast_tanh(acc[j]);
            out[gi] = v;
            const float carried = v + NSTD * nz[j];
            Hw[m * HPAD + col] = (__bf16)carried;
            if (LAST)   // compile-time: only in the peeled final step
                out[(size_t)T_STEPS * BH + (size_t)(bbase + m) * HID + col] = carried;
        }

        wg_barrier_signal();                       // publish Hw

        if (!LAST) {
            // Issue next step's global traffic while the barrier resolves.
            // Ax / nz are dead here -> refill in place (one-step pipeline).
            const int nt = t + 1;
            load_x_tile(x + ((size_t)nt * BATCH + bbase + arow) * IN, half, Ax);
            load_noise(noise + (size_t)nt * BH + (size_t)(bbase + 8 * half) * HID + col,
                       nz);
            // distance-2 prefetch (global_prefetch_b8), clamped at T-1
            const int pt = (t + 2 < T_STEPS) ? (t + 2) : (T_STEPS - 1);
            __builtin_prefetch(x + ((size_t)pt * BATCH + bbase + arow) * IN, 0, 1);
            __builtin_prefetch(noise + (size_t)pt * BH + (size_t)(bbase + arow) * HID,
                               0, 1);
        }
        p ^= 1;
    };

    for (int t = 0; t < T_STEPS - 1; ++t)
        step(t, std::false_type{});
    step(T_STEPS - 1, std::true_type{});
}

extern "C" void kernel_launch(void* const* d_in, const int* in_sizes, int n_in,
                              void* d_out, int out_size, void* d_ws, size_t ws_size,
                              hipStream_t stream) {
    (void)in_sizes; (void)n_in; (void)d_ws; (void)ws_size; (void)out_size;
    const float* x         = (const float*)d_in[0];
    const float* w_ih      = (const float*)d_in[1];
    const float* w_hh      = (const float*)d_in[2];
    const float* b_ih      = (const float*)d_in[3];
    const float* b_hh      = (const float*)d_in[4];
    const float* noise     = (const float*)d_in[5];
    const float* hidden_in = (const float*)d_in[6];
    float*       out       = (float*)d_out;

    rnn_scan_kernel<<<BATCH / 16, 256, 0, stream>>>(
        x, w_ih, w_hh, b_ih, b_hh, noise, hidden_in, out);
}
